// MambaBlock_3977139716608
// MI455X (gfx1250) — compile-verified, matrix operand order checked by reading, not run on hardware
//
#include <hip/hip_runtime.h>
#include <hip/hip_bf16.h>
#include <stdint.h>

// ---------------- problem dims ----------------
#define D_MODEL 1024
#define D_INNER 2048
#define D_STATE 16
#define DT_RANK 64
#define D_CONV  4
#define B_SZ    2
#define LSEQ    2048
#define BL      (B_SZ * LSEQ)             // 4096 token rows
#define XDBL_W  (DT_RANK + 2 * D_STATE)   // 96

// ---------------- CDNA5 WMMA types ----------------
typedef __attribute__((ext_vector_type(16))) __bf16 v16bf;
typedef __attribute__((ext_vector_type(8)))  float  v8f;

union FragU { uint4 q[2]; v16bf v; };     // 32B fragment = 2x16B LDS loads

// ---------------- CDNA5 async global->LDS path (probe-guarded) ----------------
// Builtin signature (from clang diagnostic): param1 = int4-vector ptr in AS(1)
// (printed as "__device__"), param2 = AS(3) LDS ptr, then imm offset + imm cpol.
typedef int v4i_vec __attribute__((__vector_size__(16)));
typedef __attribute__((address_space(1))) v4i_vec as1_v4i;
typedef __attribute__((address_space(3))) v4i_vec as3_v4i;

#if defined(__has_builtin)
#  if __has_builtin(__builtin_amdgcn_global_load_async_to_lds_b128)
#    define HAVE_ASYNC_B128 1
#  endif
#  if __has_builtin(__builtin_amdgcn_s_wait_asynccnt)
#    define HAVE_WAIT_ASYNC 1
#  endif
#endif

__device__ __forceinline__ void async_copy16(const unsigned short* gsrc,
                                             unsigned short* lds_dst) {
#if defined(HAVE_ASYNC_B128)
  __builtin_amdgcn_global_load_async_to_lds_b128(
      (as1_v4i*)gsrc, (as3_v4i*)lds_dst, 0, 0);
#else
  *(uint4*)lds_dst = *(const uint4*)gsrc;
#endif
}

__device__ __forceinline__ void wait_async_lds() {
#if defined(HAVE_ASYNC_B128)
#  if defined(HAVE_WAIT_ASYNC)
  __builtin_amdgcn_s_wait_asynccnt(0);
#  else
  asm volatile("s_wait_asynccnt 0x0" ::: "memory");
#  endif
#endif
}

__device__ __forceinline__ unsigned short f32_to_bf16(float f) {
  union { float f; unsigned int u; } c; c.f = f;
  unsigned int u = c.u;
  u += 0x7FFFu + ((u >> 16) & 1u);        // round-to-nearest-even
  return (unsigned short)(u >> 16);
}

// ---------------- generic cast f32 -> bf16 ----------------
__global__ void cast_f32_bf16_kernel(const float* __restrict__ in,
                                     unsigned short* __restrict__ out, int n) {
  int i = blockIdx.x * blockDim.x + threadIdx.x;
  if (i < n) out[i] = f32_to_bf16(in[i]);
}

// ---------------- bf16 WMMA GEMM: C[M,N] = A[M,K] * B[K,N] ----------------
// block = 256 threads (8 waves), workgroup tile 128x128, wave tile 64x32 (4x2 wmma).
// Double-buffered LDS; A tile staged with global_load_async_to_lds_b128 (ASYNCcnt),
// B tile prefetched to registers before the WMMA burst, scattered (transposed) after.
// Requires: M % 128 == 0, K % 32 == 0. N guarded (zero-padded tiles).
__global__ __launch_bounds__(256) void gemm_bf16_wmma(
    const unsigned short* __restrict__ A, int lda,
    const unsigned short* __restrict__ B, int ldb,
    float* __restrict__ C, int ldc,
    int M, int N, int K)
{
  __shared__ __align__(16) unsigned short As[2][128][40];  // [m][k], +8 pad (16B align)
  __shared__ __align__(16) unsigned short Bs[2][128][40];  // transposed: [n][k]

  const int tid  = threadIdx.x;
  const int lane = tid & 31;
  const int wv   = tid >> 5;
  const int m0   = blockIdx.y * 128;
  const int n0   = blockIdx.x * 128;
  const int mw   = (wv >> 2) * 64;        // 2 waves along M
  const int nw   = (wv & 3) * 32;         // 4 waves along N
  const int frow = lane & 15;             // row (A) / col (B) within 16-wide tile
  const int kb   = (lane & 16) >> 1;      // 0 or 8: K-subgroup per ISA 16-bit layout

  // per-thread staging coordinates (2 chunks each, fully unrolled)
  const int ca_r = tid >> 2;              // A rows: ca_r and ca_r+64
  const int ca_c = (tid & 3) << 3;        // A col offset (x8 halves)
  const int cb_k = tid >> 4;              // B k-rows: cb_k and cb_k+16
  const int cb_n = (tid & 15) << 3;       // B col offset (x8 halves)

  v8f acc[4][2];
  for (int t = 0; t < 4; ++t)
    for (int u = 0; u < 2; ++u)
      for (int r = 0; r < 8; ++r) acc[t][u][r] = 0.f;

  uint4 breg0, breg1;

  auto stageA = [&](int k0, int bufi) {
    async_copy16(A + (size_t)(m0 + ca_r) * lda + k0 + ca_c,        &As[bufi][ca_r][ca_c]);
    async_copy16(A + (size_t)(m0 + ca_r + 64) * lda + k0 + ca_c,   &As[bufi][ca_r + 64][ca_c]);
  };
  auto loadB = [&](int k0) {
    int gn = n0 + cb_n;
    if (gn + 8 <= N) {
      breg0 = *(const uint4*)(B + (size_t)(k0 + cb_k) * ldb + gn);
      breg1 = *(const uint4*)(B + (size_t)(k0 + cb_k + 16) * ldb + gn);
    } else {
      unsigned short t0[8], t1[8];
      for (int j = 0; j < 8; ++j) {
        t0[j] = (gn + j < N) ? B[(size_t)(k0 + cb_k) * ldb + gn + j] : (unsigned short)0;
        t1[j] = (gn + j < N) ? B[(size_t)(k0 + cb_k + 16) * ldb + gn + j] : (unsigned short)0;
      }
      breg0 = *(uint4*)t0; breg1 = *(uint4*)t1;
    }
  };
  auto scatterB = [&](int bufi) {
    unsigned short t0[8], t1[8];
    *(uint4*)t0 = breg0; *(uint4*)t1 = breg1;
    for (int j = 0; j < 8; ++j) {
      Bs[bufi][cb_n + j][cb_k]      = t0[j];
      Bs[bufi][cb_n + j][cb_k + 16] = t1[j];
    }
  };

  // prologue: stage tile 0
  int buf = 0;
  stageA(0, 0);
  loadB(0);
  scatterB(0);
  wait_async_lds();
  __syncthreads();

  for (int k0 = 0; k0 < K; k0 += 32) {
    const int  nxt  = buf ^ 1;
    const bool more = (k0 + 32) < K;
    if (more) {               // issue next tile while this one computes
      stageA(k0 + 32, nxt);   // async -> LDS, tracked by ASYNCcnt
      loadB(k0 + 32);         // global -> regs, latency hidden by WMMAs
    }

    // --- fragments per ISA VGPR layout (two contiguous 16B groups per lane) ---
    FragU fa[4], fb[2];
    for (int t = 0; t < 4; ++t) {
      fa[t].q[0] = *(const uint4*)&As[buf][mw + t * 16 + frow][kb];
      fa[t].q[1] = *(const uint4*)&As[buf][mw + t * 16 + frow][kb + 16];
    }
    for (int u = 0; u < 2; ++u) {
      fb[u].q[0] = *(const uint4*)&Bs[buf][nw + u * 16 + frow][kb];
      fb[u].q[1] = *(const uint4*)&Bs[buf][nw + u * 16 + frow][kb + 16];
    }
    for (int t = 0; t < 4; ++t)
      for (int u = 0; u < 2; ++u)
        acc[t][u] = __builtin_amdgcn_wmma_f32_16x16x32_bf16(
            false, fa[t].v, false, fb[u].v, (short)0, acc[t][u], false, false);

    if (more) scatterB(nxt);  // regs -> LDS (other buffer, no race with readers of buf)
    wait_async_lds();         // drain this wave's async A-tile writes
    __syncthreads();          // all waves' tile 'nxt' complete & 'buf' fully consumed
    buf = nxt;
  }

  // --- store: C/D layout: M = r + 8*(lane>=16), N = lane&15 ---
  const int rowoff = (lane & 16) >> 1;
  for (int t = 0; t < 4; ++t)
    for (int u = 0; u < 2; ++u) {
      int gn = n0 + nw + u * 16 + frow;
      if (gn < N) {
        for (int r = 0; r < 8; ++r) {
          int gm = m0 + mw + t * 16 + r + rowoff;
          C[(size_t)gm * ldc + gn] = acc[t][u][r];
        }
      }
    }
}

// ---------------- causal depthwise conv (k=4) + bias + silu ----------------
__global__ void conv_silu_kernel(const float* __restrict__ XR,
                                 const float* __restrict__ conv_w,
                                 const float* __restrict__ conv_b,
                                 float* __restrict__ xi_f,
                                 unsigned short* __restrict__ xi_b) {
  int idx = blockIdx.x * blockDim.x + threadIdx.x;   // (b*L + l)*D_INNER + d
  if (idx >= BL * D_INNER) return;
  int d  = idx & (D_INNER - 1);
  int bl = idx >> 11;
  int l  = bl & (LSEQ - 1);
  int b  = bl >> 11;
  float acc = conv_b[d];
#pragma unroll
  for (int k = 0; k < D_CONV; ++k) {
    int ll = l - (D_CONV - 1) + k;
    if (ll >= 0)
      acc += conv_w[d * D_CONV + k] *
             XR[(size_t)(b * LSEQ + ll) * (2 * D_INNER) + d];
  }
  float s = acc / (1.f + __expf(-acc));   // silu
  xi_f[idx] = s;
  xi_b[idx] = f32_to_bf16(s);
}

// ---------------- slice x_dbl[:, :64] -> bf16 ----------------
__global__ void dpart_cast_kernel(const float* __restrict__ xdbl,
                                  unsigned short* __restrict__ dpb) {
  int i = blockIdx.x * blockDim.x + threadIdx.x;   // BL * DT_RANK
  if (i >= BL * DT_RANK) return;
  int r = i >> 6, c = i & 63;
  dpb[i] = f32_to_bf16(xdbl[r * XDBL_W + c]);
}

// ---------------- + dt_proj_b, softplus -> bf16 ----------------
__global__ void softplus_kernel(const float* __restrict__ dtraw,
                                const float* __restrict__ bias,
                                unsigned short* __restrict__ delta_b) {
  int i = blockIdx.x * blockDim.x + threadIdx.x;   // BL * D_INNER
  if (i >= BL * D_INNER) return;
  float v  = dtraw[i] + bias[i & (D_INNER - 1)];
  float sp = (v > 20.f) ? v : log1pf(__expf(v));
  delta_b[i] = f32_to_bf16(sp);
}

// ---------------- fused selective scan ----------------
// one wave = 2 channels: lanes 0-15 hold states n=0..15 of d, lanes 16-31 of d+1.
// deltaA / dBu computed on the fly (never materialized: saves ~1GB of HBM traffic).
// epilogue fuses  y = y + xi*D  and  y *= silu(res),  writes bf16 for out_proj GEMM.
__global__ __launch_bounds__(256) void ssm_scan_kernel(
    const float* __restrict__ delta_p,   // [BL][D_INNER]
    const float* __restrict__ xi_f,      // [BL][D_INNER]
    const float* __restrict__ xdbl,      // [BL][96]  (B @ 64..79, C @ 80..95)
    const float* __restrict__ A_log,     // [D_INNER][16]
    const float* __restrict__ Dvec,      // [D_INNER]
    const float* __restrict__ XR,        // [BL][2*D_INNER] (res at cols 2048..)
    unsigned short* __restrict__ yact_b) // [BL][D_INNER] bf16
{
  const int lane  = threadIdx.x & 31;
  const int gw    = (blockIdx.x * blockDim.x + threadIdx.x) >> 5;  // 0..2047
  const int b     = gw >> 10;                // / (D_INNER/2)
  const int dpair = gw & 1023;
  const int d     = (dpair << 1) | (lane >> 4);
  const int n     = lane & 15;

  const float Aval = -__expf(A_log[d * D_STATE + n]);
  const float Dd   = Dvec[d];
  float h = 0.f;

  for (int l = 0; l < LSEQ; ++l) {
    const size_t base = (size_t)(b * LSEQ + l);
    float dp = delta_p[base * D_INNER + d];        // broadcast within half-wave
    float xv = xi_f[base * D_INNER + d];
    float Bn = xdbl[base * XDBL_W + DT_RANK + n];
    float Cn = xdbl[base * XDBL_W + DT_RANK + D_STATE + n];

    h = __expf(dp * Aval) * h + dp * Bn * xv;      // h = dA*h + dBu
    float y = h * Cn;
#pragma unroll
    for (int off = 8; off > 0; off >>= 1) y += __shfl_xor(y, off, 16);

    if (n == 0) {
      float resv = XR[base * (2 * D_INNER) + D_INNER + d];
      float sres = resv / (1.f + __expf(-resv));   // silu(res)
      yact_b[base * D_INNER + d] = f32_to_bf16((y + xv * Dd) * sres);
    }
  }
}

// ---------------- orchestration ----------------
extern "C" void kernel_launch(void* const* d_in, const int* in_sizes, int n_in,
                              void* d_out, int out_size, void* d_ws, size_t ws_size,
                              hipStream_t stream) {
  const float* x    = (const float*)d_in[0];   // [2,2048,1024]
  const float* adj  = (const float*)d_in[1];   // [2048,2048]
  const float* ipw  = (const float*)d_in[2];   // [1024,4096]
  const float* cw   = (const float*)d_in[3];   // [2048,1,4]
  const float* cb   = (const float*)d_in[4];   // [2048]
  const float* xpw  = (const float*)d_in[5];   // [2048,96]
  const float* dtw  = (const float*)d_in[6];   // [64,2048]
  const float* dtb  = (const float*)d_in[7];   // [2048]
  const float* Alog = (const float*)d_in[8];   // [2048,16]
  const float* Dv   = (const float*)d_in[9];   // [2048]
  const float* opw  = (const float*)d_in[10];  // [2048,1024]
  float* out = (float*)d_out;                  // [2,2048,1024] fp32

  // carve workspace (256B aligned regions)
  char* p = (char*)d_ws;
  auto carve = [&](size_t bytes) -> void* {
    void* r = (void*)p;
    p += (bytes + 255) & ~(size_t)255;
    return r;
  };
  unsigned short* Xb     = (unsigned short*)carve((size_t)BL * D_MODEL * 2);
  unsigned short* W1b    = (unsigned short*)carve((size_t)D_MODEL * 2 * D_INNER * 2);
  unsigned short* adjb   = (unsigned short*)carve((size_t)D_INNER * D_INNER * 2);
  unsigned short* xpwb   = (unsigned short*)carve((size_t)D_INNER * XDBL_W * 2);
  unsigned short* dtwb   = (unsigned short*)carve((size_t)DT_RANK * D_INNER * 2);
  unsigned short* opwb   = (unsigned short*)carve((size_t)D_INNER * D_MODEL * 2);
  float*          XR     = (float*)carve((size_t)BL * 2 * D_INNER * 4);
  float*          xi_f   = (float*)carve((size_t)BL * D_INNER * 4);
  unsigned short* xi_b   = (unsigned short*)carve((size_t)BL * D_INNER * 2);
  float*          xdbl   = (float*)carve((size_t)BL * XDBL_W * 4);
  unsigned short* dpb    = (unsigned short*)carve((size_t)BL * DT_RANK * 2);
  float*          dtraw  = (float*)carve((size_t)BL * D_INNER * 4);
  unsigned short* deltab = (unsigned short*)carve((size_t)BL * D_INNER * 2);
  float*          dP     = (float*)carve((size_t)BL * D_INNER * 4);
  unsigned short* yactb  = (unsigned short*)carve((size_t)BL * D_INNER * 2);

  const int T = 256;
  auto blks = [](int n) { return (n + 255) / 256; };

  // 1. casts to bf16
  cast_f32_bf16_kernel<<<blks(BL * D_MODEL), T, 0, stream>>>(x, Xb, BL * D_MODEL);
  cast_f32_bf16_kernel<<<blks(D_MODEL * 2 * D_INNER), T, 0, stream>>>(ipw, W1b, D_MODEL * 2 * D_INNER);
  cast_f32_bf16_kernel<<<blks(D_INNER * D_INNER), T, 0, stream>>>(adj, adjb, D_INNER * D_INNER);
  cast_f32_bf16_kernel<<<blks(D_INNER * XDBL_W), T, 0, stream>>>(xpw, xpwb, D_INNER * XDBL_W);
  cast_f32_bf16_kernel<<<blks(DT_RANK * D_INNER), T, 0, stream>>>(dtw, dtwb, DT_RANK * D_INNER);
  cast_f32_bf16_kernel<<<blks(D_INNER * D_MODEL), T, 0, stream>>>(opw, opwb, D_INNER * D_MODEL);

  // 2. in_proj: XR[4096,4096] = Xb[4096,1024] @ W1b[1024,4096]
  gemm_bf16_wmma<<<dim3((2 * D_INNER) / 128, BL / 128), T, 0, stream>>>(
      Xb, D_MODEL, W1b, 2 * D_INNER, XR, 2 * D_INNER, BL, 2 * D_INNER, D_MODEL);

  // 3. depthwise conv + silu -> xi (fp32 + bf16)
  conv_silu_kernel<<<blks(BL * D_INNER), T, 0, stream>>>(XR, cw, cb, xi_f, xi_b);

  // 4. x_proj: xdbl[4096,96] = xi @ xpw
  gemm_bf16_wmma<<<dim3(1, BL / 128), T, 0, stream>>>(
      xi_b, D_INNER, xpwb, XDBL_W, xdbl, XDBL_W, BL, XDBL_W, D_INNER);

  // 5. delta path: slice->bf16, dt_proj GEMM, +bias softplus -> bf16
  dpart_cast_kernel<<<blks(BL * DT_RANK), T, 0, stream>>>(xdbl, dpb);
  gemm_bf16_wmma<<<dim3(D_INNER / 128, BL / 128), T, 0, stream>>>(
      dpb, DT_RANK, dtwb, D_INNER, dtraw, D_INNER, BL, D_INNER, DT_RANK);
  softplus_kernel<<<blks(BL * D_INNER), T, 0, stream>>>(dtraw, dtb, deltab);

  // 6. adjacency modulation: dP[4096,2048] = delta @ adj
  gemm_bf16_wmma<<<dim3(D_INNER / 128, BL / 128), T, 0, stream>>>(
      deltab, D_INNER, adjb, D_INNER, dP, D_INNER, BL, D_INNER, D_INNER);

  // 7. fused selective scan (+ D skip + silu(res) gate) -> yact bf16
  ssm_scan_kernel<<<256, T, 0, stream>>>(dP, xi_f, xdbl, Alog, Dv, XR, yactb);

  // 8. out_proj: out[4096,1024] = yact @ opw
  gemm_bf16_wmma<<<dim3(D_MODEL / 128, BL / 128), T, 0, stream>>>(
      yactb, D_INNER, opwb, D_MODEL, out, D_MODEL, BL, D_MODEL, D_INNER);
}
